// CRF_76398878261444
// MI455X (gfx1250) — matrix operand order repaired and checked by reading, not compiled, not run on hardware
//
#include <hip/hip_runtime.h>

#define LDIM  256   // number of labels L
#define TLEN  512   // sequence length T
#define BTOT  128   // total batch B
#define BTILE 16    // batches per workgroup (one WMMA N-tile)
#define EASTR 264   // padded row stride (halves) for transposed expA in LDS

typedef _Float16 v16h __attribute__((ext_vector_type(16)));
typedef _Float16 v8h  __attribute__((ext_vector_type(8)));
typedef float    v8f  __attribute__((ext_vector_type(8)));

__global__ void crf_zero_out(float* out) { out[0] = 0.0f; }

__global__ __launch_bounds__(256, 1)
void crf_forward_kernel(const float* __restrict__ emis,
                        const float* __restrict__ trans,
                        const int*   __restrict__ words,
                        float*       __restrict__ out)
{
    __shared__ float    sAlpha[LDIM * BTILE];   // alpha[j][b], 16 KB
    __shared__ _Float16 sExpAT[BTILE * EASTR];  // exp(alpha-Am)[b][k], padded, 8.25 KB
    __shared__ float    sTm[LDIM];              // row max of trans
    __shared__ float    sAm[BTILE];             // per-batch alpha max
    __shared__ float    sRed[256];              // reduction scratch

    const int tid  = threadIdx.x;
    const int lane = tid & 31;       // wave32 lane
    const int wv   = tid >> 5;       // wave id 0..7 -> M rows [wv*32, wv*32+32)
    const int lrow = lane & 15;      // A: M row within tile / B: N column (batch)
    const int lhi  = lane >> 4;      // half-wave select
    const int b0   = blockIdx.x * BTILE;

    // ---- Tm[i] = max_j trans[i][j] (constant over all time steps) ----
    {
        float m = -3.4e38f;
        const float* row = trans + tid * LDIM;
        for (int j = 0; j < LDIM; ++j) m = fmaxf(m, row[j]);
        sTm[tid] = m;
    }

    // ---- alpha0[i][b] = emis[words[b][0]][i] ----
    for (int idx = tid; idx < LDIM * BTILE; idx += 256) {
        const int i = idx >> 4, b = idx & 15;
        const int w0 = words[(b0 + b) * TLEN];
        sAlpha[idx] = emis[(size_t)w0 * LDIM + i];
    }
    __syncthreads();

    // ---- A fragments: exp(trans[i][k] - Tm[i]) in f16, register-resident ----
    // 16-bit A 16x32 layout: lanes 0-15 M=0..15, K = e + (e>=8)*8; lanes 16-31 K += 8
    v16h aF[2][8];
    #pragma unroll
    for (int mt = 0; mt < 2; ++mt) {
        const int iRow = (wv * 2 + mt) * 16 + lrow;
        const float tm = sTm[iRow];
        const float* trow = trans + iRow * LDIM;
        #pragma unroll
        for (int kt = 0; kt < 8; ++kt) {
            v16h f;
            #pragma unroll
            for (int e = 0; e < 16; ++e) {
                const int k = kt * 32 + e + (lhi ? 8 : 0) + (e >= 8 ? 8 : 0);
                f[e] = (_Float16)__expf(trow[k] - tm);
            }
            aF[mt][kt] = f;
        }
    }

    const int bloc  = lrow;
    const int bglob = b0 + bloc;

    #pragma unroll 1
    for (int t = 1; t < TLEN; ++t) {
        __syncthreads();                       // prior-step alpha writes visible

        // ---- Am[b] = max_j alpha[j][b] ----
        {
            const int b = tid & 15, g = tid >> 4;
            float m = -3.4e38f;
            #pragma unroll
            for (int k = 0; k < 16; ++k) m = fmaxf(m, sAlpha[(g * 16 + k) * 16 + b]);
            sRed[g * 16 + b] = m;
        }
        __syncthreads();
        if (tid < 16) {
            float m = -3.4e38f;
            #pragma unroll
            for (int g = 0; g < 16; ++g) m = fmaxf(m, sRed[g * 16 + tid]);
            sAm[tid] = m;
        }
        __syncthreads();

        // ---- expA[b][k] = exp(alpha[k][b] - Am[b]) (f16, transposed + padded) ----
        for (int idx = tid; idx < LDIM * BTILE; idx += 256) {
            const int k = idx >> 4, b = idx & 15;
            sExpAT[b * EASTR + k] = (_Float16)__expf(sAlpha[idx] - sAm[b]);
        }
        __syncthreads();

        // ---- GEMM: acc[i,b] = sum_j expT[i,j] * expA[j,b] via WMMA f16->f32 ----
        // Dense B 32x16 layout: lane = N (batch), K = e + (e>=8)*8; lanes 16-31 K += 8
        v8f acc0 = {};
        v8f acc1 = {};
        #pragma unroll
        for (int kt = 0; kt < 8; ++kt) {
            const _Float16* p = &sExpAT[lrow * EASTR + kt * 32 + lhi * 8];
            const v8h lo = *(const v8h*)p;
            const v8h hi = *(const v8h*)(p + 16);
            v16h bF;
            #pragma unroll
            for (int e = 0; e < 8; ++e) { bF[e] = lo[e]; bF[e + 8] = hi[e]; }
            acc0 = __builtin_amdgcn_wmma_f32_16x16x32_f16(false, aF[0][kt], false, bF,
                                                          (short)0, acc0, false, false);
            acc1 = __builtin_amdgcn_wmma_f32_16x16x32_f16(false, aF[1][kt], false, bF,
                                                          (short)0, acc1, false, false);
        }

        // ---- epilogue: alpha_new[i][b] = e_t + Tm[i] + Am[b] + log(acc) ----
        const int w = words[bglob * TLEN + t];
        const float* erow = emis + (size_t)w * LDIM;
        const float amb = sAm[bloc];
        #pragma unroll
        for (int mt = 0; mt < 2; ++mt) {
            const int ibase = (wv * 2 + mt) * 16 + lhi * 8;
            #pragma unroll
            for (int r = 0; r < 8; ++r) {
                const int i = ibase + r;
                const float v = (mt ? acc1[r] : acc0[r]);
                sAlpha[i * 16 + bloc] = erow[i] + sTm[i] + amb + __logf(v);
            }
        }
    }

    // ---- log_Z contribution: sum_b logsumexp_i alpha_last[i][b] ----
    __syncthreads();
    {
        const int b = tid & 15, g = tid >> 4;
        float m = -3.4e38f;
        #pragma unroll
        for (int k = 0; k < 16; ++k) m = fmaxf(m, sAlpha[(g * 16 + k) * 16 + b]);
        sRed[g * 16 + b] = m;
    }
    __syncthreads();
    if (tid < 16) {
        float m = -3.4e38f;
        #pragma unroll
        for (int g = 0; g < 16; ++g) m = fmaxf(m, sRed[g * 16 + tid]);
        sAm[tid] = m;
    }
    __syncthreads();
    {
        const int b = tid & 15, g = tid >> 4;
        float s = 0.0f;
        #pragma unroll
        for (int k = 0; k < 16; ++k) s += __expf(sAlpha[(g * 16 + k) * 16 + b] - sAm[b]);
        sRed[g * 16 + b] = s;
    }
    __syncthreads();
    if (tid < 16) {
        float s = 0.0f;
        #pragma unroll
        for (int g = 0; g < 16; ++g) s += sRed[g * 16 + tid];
        sRed[tid] = sAm[tid] + __logf(s);
    }
    __syncthreads();
    if (tid == 0) {
        float tot = 0.0f;
        for (int b = 0; b < BTILE; ++b) tot += sRed[b];
        atomicAdd(out, tot);
    }
}

extern "C" void kernel_launch(void* const* d_in, const int* in_sizes, int n_in,
                              void* d_out, int out_size, void* d_ws, size_t ws_size,
                              hipStream_t stream)
{
    const float* emis  = (const float*)d_in[0];   // (V+1, L) f32
    const float* trans = (const float*)d_in[1];   // (L, L)   f32
    const int*   words = (const int*)d_in[2];     // (B, T)   i32
    float* out = (float*)d_out;                   // scalar f32

    hipLaunchKernelGGL(crf_zero_out, dim3(1), dim3(1), 0, stream, out);
    hipLaunchKernelGGL(crf_forward_kernel, dim3(BTOT / BTILE), dim3(256), 0, stream,
                       emis, trans, words, out);
}
